// MTL4CharCNNWordBilstmModel_51625506898624
// MI455X (gfx1250) — matrix operand order repaired and checked by reading, not compiled, not run on hardware
//
#include <hip/hip_runtime.h>

// ---------------- problem constants ----------------
#define Bq     32
#define Tq     512
#define Dq     512
#define Hq     512
#define G4     2048          // 4*H
#define KD     1024          // D+H (weight rows)
#define NDIR   10            // 5 forward + 5 backward LSTMs
#define NSPLIT 8             // workgroups per direction in the recurrence

// ---------------- workspace layout (bytes) ----------------
#define WS_CTR_OFF  ((size_t)0)                              // 10*512 ints = 20,480 B
#define WS_HBUF_OFF ((size_t)32768)                          // 2*10*32*512 bf16 = 655,360 B (double-buffered)
#define WS_XP_OFF   ((size_t)688128)                         // 10*16384*2048 bf16 = 671,088,640 B
#define WS_OUT_OFF  (WS_XP_OFF + (size_t)NDIR*Bq*Tq*G4*2)    // 10*16384*512 bf16 = 167,772,160 B

#define K2_SMEM (262144 /*Wh*/ + 32768 /*h*/ + 16384 /*xp*/ + 512 /*idx/mask/len*/)

typedef float  v8f   __attribute__((ext_vector_type(8)));
typedef __bf16 v16bf __attribute__((ext_vector_type(16)));
typedef __bf16 v8bf  __attribute__((ext_vector_type(8)));

// native converts (gfx1250 has bf16 VALU cvt; RNE matches reference rounding intent)
static __device__ __forceinline__ __bf16 f2bf(float f) { return (__bf16)f; }
static __device__ __forceinline__ float bf2f(__bf16 b) { return (float)b; }
static __device__ __forceinline__ float sigm(float x) { return 1.0f / (1.0f + __expf(-x)); }

static __device__ __forceinline__ v16bf cat16(v8bf lo, v8bf hi) {
  return __builtin_shufflevector(lo, hi, 0,1,2,3,4,5,6,7,8,9,10,11,12,13,14,15);
}
static __device__ __forceinline__ v8f wmma_bf16(v16bf a, v16bf b, v8f c) {
  return __builtin_amdgcn_wmma_f32_16x16x32_bf16(false, a, false, b, (short)0, c, false, false);
}
// A fragment (16x32 bf16): lane m=lane&15 holds K block [kh*16, kh*16+16)
static __device__ __forceinline__ v16bf load_afrag(const __bf16* row /* &A[m][0] */, int kh) {
  v8bf lo = *(const v8bf*)(row + kh * 16);
  v8bf hi = *(const v8bf*)(row + kh * 16 + 8);
  return cat16(lo, hi);
}
static __device__ __forceinline__ v8f zero8() {
  v8f z = {0.f,0.f,0.f,0.f,0.f,0.f,0.f,0.f};
  return z;
}

// ---------------- kernel 0: zero the per-(dir,step) barrier counters ----------------
__global__ void k0_init(int* __restrict__ ctr) {
  int i = blockIdx.x * 256 + threadIdx.x;
  if (i < NDIR * Tq) ctr[i] = 0;
}

// ---------------- kernel 1: XP[w] = X @ Wx[w] + b[w]  (bf16 out) ----------------
// block tile 64(M) x 64(N), 8 waves, 2 C-tiles per wave (B-frag reused), K = 512 in 16 chunks.
__global__ __launch_bounds__(256) void k1_xproj(
    const float* __restrict__ X,  const float* __restrict__ Wfw, const float* __restrict__ bfw,
    const float* __restrict__ Wbw, const float* __restrict__ bbw, __bf16* __restrict__ XP) {
  __shared__ __bf16 As[64 * 32];
  __shared__ __bf16 Bs[64 * 32];   // transposed: Bs[n][k]

  const int tid  = threadIdx.x;
  const int lane = tid & 31, wv = tid >> 5;
  const int mh = wv & 1, nt = wv >> 1;       // M-half (32 rows) x 4 N-tiles
  const int n16 = lane & 15, hl = lane >> 4;

  const int mblk = blockIdx.x, nblk = blockIdx.y, w = blockIdx.z;
  const float* Wsel = (w < 5) ? (Wfw + (size_t)w * KD * G4) : (Wbw + (size_t)(w - 5) * KD * G4);
  const float* bsel = (w < 5) ? (bfw + (size_t)w * G4)      : (bbw + (size_t)(w - 5) * G4);
  const int r0 = mblk * 64, n0 = nblk * 64;

  v8f acc0 = zero8(), acc1 = zero8();
  for (int kc = 0; kc < 16; ++kc) {
    const int k0 = kc * 32;
    __syncthreads();
#pragma unroll
    for (int i = 0; i < 8; ++i) {            // As: 64x32 from X (fp32 -> bf16)
      int flat = i * 256 + tid, m = flat >> 5, kk = flat & 31;
      As[m * 32 + kk] = f2bf(X[(size_t)(r0 + m) * Dq + k0 + kk]);
    }
#pragma unroll
    for (int i = 0; i < 8; ++i) {            // Bs[n][k] = Wx[k][n]
      int flat = i * 256 + tid, k = flat >> 6, n = flat & 63;
      Bs[n * 32 + k] = f2bf(Wsel[(size_t)(k0 + k) * G4 + n0 + n]);
    }
    __syncthreads();
    v16bf a0 = load_afrag(&As[(mh * 32 + n16) * 32], hl);
    v16bf a1 = load_afrag(&As[(mh * 32 + 16 + n16) * 32], hl);
    v16bf b  = *(const v16bf*)&Bs[(nt * 16 + n16) * 32 + hl * 16];
    acc0 = wmma_bf16(a0, b, acc0);
    acc1 = wmma_bf16(a1, b, acc1);
  }
  const int colg = n0 + nt * 16 + n16;
  const float bias = bsel[colg];
#pragma unroll
  for (int r = 0; r < 8; ++r) {
    int row0 = r0 + mh * 32 + hl * 8 + r;
    XP[((size_t)w * (Bq * Tq) + row0) * G4 + colg]        = f2bf(acc0[r] + bias);
    XP[((size_t)w * (Bq * Tq) + row0 + 16) * G4 + colg]   = f2bf(acc1[r] + bias);
  }
}

// ---------------- kernel 2: LSTM recurrence ----------------
// grid = 10 dirs * 8 splits. Each WG owns 64 h-columns (=> 256 z-columns across i,j,f,o).
// Wh slice (512x256 bf16 = 256KB) resident in LDS for all 512 steps. h broadcast through
// a DOUBLE-BUFFERED global h staging area (skew across the step barrier is at most 1 step).
__global__ __launch_bounds__(256) void k2_lstm(
    const float* __restrict__ Wfw, const float* __restrict__ Wbw,
    const int* __restrict__ lens, const __bf16* __restrict__ XP,
    __bf16* __restrict__ OUT, __bf16* __restrict__ HBUF, int* __restrict__ CTR) {
  extern __shared__ char smem[];
  __bf16* Wh_s = (__bf16*)smem;                              // [256][512]
  __bf16* h_s  = (__bf16*)(smem + 262144);                   // [32][512]
  __bf16* xp_s = (__bf16*)(smem + 262144 + 32768);           // [32][256]  (b, g*64+c)
  int*    tix_s = (int*)  (smem + 262144 + 32768 + 16384);   // [32]
  float*  msk_s = (float*)(smem + 262144 + 32768 + 16384 + 128);
  int*    len_s = (int*)  (smem + 262144 + 32768 + 16384 + 256);

  const int tid  = threadIdx.x;
  const int lane = tid & 31, wv = tid >> 5;
  const int mt = wv & 1, nt = wv >> 1;            // M-tile (batch), col-tile within 64
  const int n16 = lane & 15, hl = lane >> 4;

  const int dir = blockIdx.x / NSPLIT, split = blockIdx.x % NSPLIT;
  const float* Wsel = (dir < 5) ? (Wfw + (size_t)dir * KD * G4) : (Wbw + (size_t)(dir - 5) * KD * G4);

  // ---- preload Wh slice into LDS, transposed: Wh_s[nl][k] with nl = g*64 + c ----
  {
    const int g = tid >> 6, c = tid & 63;
    const size_t col = (size_t)g * Hq + (size_t)split * 64 + c;   // z-column in [0,2048)
    for (int k = 0; k < Hq; ++k)
      Wh_s[(size_t)tid * Hq + k] = f2bf(Wsel[(size_t)(Dq + k) * G4 + col]);
  }
  unsigned* h32 = (unsigned*)h_s;
  for (int i = 0; i < 32; ++i) h32[i * 256 + tid] = 0u;           // h = 0
  if (tid < Bq) len_s[tid] = lens[tid];
  __syncthreads();

  float c_reg[8], h_reg[8];
#pragma unroll
  for (int r = 0; r < 8; ++r) { c_reg[r] = 0.f; h_reg[r] = 0.f; }

  const __bf16* XPd = XP + (size_t)dir * (Bq * Tq) * G4;
  __bf16* OUTd = OUT + (size_t)dir * Bq * Tq * Hq;
  int* ctrd = CTR + dir * Tq;
  const int ncol = split * 64 + nt * 16 + n16;                    // owned h-column
  const int pfoff = (dir < 5) ? 4096 : -4096;                     // next-step XP row is +/-1 t

  for (int t = 0; t < Tq; ++t) {
    __bf16* HBw = HBUF + ((size_t)(t & 1) * NDIR + dir) * (Bq * Hq);  // double buffer

    if (tid < Bq) {                                               // per-batch time index + mask
      int L = len_s[tid];
      tix_s[tid] = (dir < 5) ? t : ((t < L) ? (L - 1 - t) : t);   // fw: t; bw: reversed gather
      msk_s[tid] = (t < L) ? 1.f : 0.f;
    }
    __syncthreads();

    // stage xp rows for this step: xp_s[b][g*64+c] (dword loads, uses per-b tix)
    {
      unsigned* xp32 = (unsigned*)xp_s;
#pragma unroll
      for (int i = 0; i < 16; ++i) {
        int flat = i * 256 + tid;                  // 4096 dwords
        int b = flat >> 7, lc2 = flat & 127;
        int c2 = lc2 * 2, g = c2 >> 6, c = c2 & 63;
        const unsigned* src = (const unsigned*)(XPd + ((size_t)b * Tq + tix_s[b]) * G4 +
                                                (size_t)g * Hq + (size_t)split * 64 + c);
        xp32[b * 128 + lc2] = *src;
        if ((flat & 31) == 0)                      // speculative prefetch of next step's row
          __builtin_prefetch((const char*)src + pfoff, 0, 0);
      }
    }

    // z = h @ Wh for the 4 gate tiles owned by this wave
    v8f acc[4];
#pragma unroll
    for (int g = 0; g < 4; ++g) acc[g] = zero8();
#pragma unroll 4
    for (int kc = 0; kc < 16; ++kc) {
      const __bf16* arow = h_s + (size_t)(mt * 16 + n16) * Hq + kc * 32;
      v16bf a = load_afrag(arow, hl);
#pragma unroll
      for (int g = 0; g < 4; ++g) {
        const v16bf* bp = (const v16bf*)(Wh_s + (size_t)(g * 64 + nt * 16 + n16) * Hq +
                                         kc * 32 + hl * 16);
        acc[g] = wmma_bf16(a, *bp, acc[g]);
      }
    }
    __syncthreads();   // xp_s staged; h_s reads complete

    // elementwise LSTM cell (TF gate order i,j,f,o; forget_bias = 1)
#pragma unroll
    for (int r = 0; r < 8; ++r) {
      const int b = mt * 16 + hl * 8 + r;
      const int xo = b * 256 + nt * 16 + n16;
      float zi = acc[0][r] + bf2f(xp_s[xo]);
      float zj = acc[1][r] + bf2f(xp_s[xo + 64]);
      float zf = acc[2][r] + bf2f(xp_s[xo + 128]);
      float zo = acc[3][r] + bf2f(xp_s[xo + 192]);
      float cn = sigm(zf + 1.f) * c_reg[r] + sigm(zi) * tanhf(zj);
      float hn = sigm(zo) * tanhf(cn);
      float m  = msk_s[b];
      c_reg[r] = m * cn + (1.f - m) * c_reg[r];
      h_reg[r] = m * hn + (1.f - m) * h_reg[r];
      HBw[(size_t)b * Hq + ncol] = f2bf(h_reg[r]);                      // broadcast h
      OUTd[((size_t)b * Tq + tix_s[b]) * Hq + ncol] = f2bf(m * hn);     // (bw: scatter-reverse)
    }

    // ---- grid barrier over the 8 WGs of this direction ----
    __threadfence();
    __syncthreads();
    if (tid == 0) {
      __hip_atomic_fetch_add(&ctrd[t], 1, __ATOMIC_ACQ_REL, __HIP_MEMORY_SCOPE_AGENT);
      while (__hip_atomic_load(&ctrd[t], __ATOMIC_ACQUIRE, __HIP_MEMORY_SCOPE_AGENT) < NSPLIT)
        __builtin_amdgcn_s_sleep(1);
    }
    __syncthreads();

    // reload full h into LDS for next step (from this step's buffer)
    {
      const unsigned* hb32 = (const unsigned*)HBw;
#pragma unroll 4
      for (int i = 0; i < 32; ++i) h32[i * 256 + tid] = hb32[i * 256 + tid];
    }
    __syncthreads();
  }
}

// ---------------- kernel 3: 4 FC heads over gathered features ----------------
// feat(head) K-sections: [fw_head | bw_head | fw_shared | bw_shared], each 512 wide.
__global__ __launch_bounds__(256) void k3_heads(
    const __bf16* __restrict__ OUT,
    const float* __restrict__ W0, const float* __restrict__ B0,
    const float* __restrict__ W1, const float* __restrict__ B1,
    const float* __restrict__ W2, const float* __restrict__ B2,
    const float* __restrict__ W3, const float* __restrict__ B3,
    float* __restrict__ Y) {
  __shared__ __bf16 As[128 * 32];
  __shared__ __bf16 Bs[32 * 32];

  const int hd = blockIdx.y;
  const int tags[4] = {17, 9, 25, 13};
  const int coffs[4] = {0, 17, 26, 51};
  const float* Wh[4] = {W0, W1, W2, W3};
  const float* Bh[4] = {B0, B1, B2, B3};
  const float* Wk = Wh[hd]; const float* bk = Bh[hd];
  const int th = tags[hd], coff = coffs[hd];
  const int dmap[4] = {hd, 5 + hd, 4, 9};

  const int tid = threadIdx.x, lane = tid & 31, wv = tid >> 5;
  const int n16 = lane & 15, hl = lane >> 4;
  const int r0 = blockIdx.x * 128;

  v8f acc[2];
  acc[0] = zero8(); acc[1] = zero8();

  for (int kc = 0; kc < 64; ++kc) {
    const int k0 = kc * 32;
    const int q = k0 >> 9, koff = k0 & 511;
    const int dq = dmap[q];
    __syncthreads();
#pragma unroll
    for (int i = 0; i < 16; ++i) {            // As: 128 rows x 32 k from OUT[dq]
      int flat = i * 256 + tid, row = flat >> 5, kk = flat & 31;
      int rg = r0 + row;                       // = b*T + t
      As[row * 32 + kk] = OUT[((size_t)dq * (Bq * Tq) + rg) * Hq + koff + kk];
    }
#pragma unroll
    for (int i = 0; i < 4; ++i) {             // Bs[n][k], zero-padded to 32 cols
      int flat = i * 256 + tid, n = flat >> 5, kk = flat & 31;
      float v = (n < th) ? Wk[(size_t)(k0 + kk) * th + n] : 0.f;
      Bs[n * 32 + kk] = f2bf(v);
    }
    __syncthreads();
    v16bf a = load_afrag(&As[(wv * 16 + n16) * 32], hl);
#pragma unroll
    for (int ntile = 0; ntile < 2; ++ntile) {
      const v16bf* bp = (const v16bf*)&Bs[(ntile * 16 + n16) * 32 + hl * 16];
      acc[ntile] = wmma_bf16(a, *bp, acc[ntile]);
    }
  }
#pragma unroll
  for (int ntile = 0; ntile < 2; ++ntile) {
    int col = ntile * 16 + n16;
    if (col < th) {
      float bv = bk[col];
#pragma unroll
      for (int r = 0; r < 8; ++r) {
        int rg = r0 + wv * 16 + hl * 8 + r;
        Y[(size_t)rg * 64 + coff + col] = acc[ntile][r] + bv;
      }
    }
  }
}

// ---------------- host ----------------
extern "C" void kernel_launch(void* const* d_in, const int* in_sizes, int n_in,
                              void* d_out, int out_size, void* d_ws, size_t ws_size,
                              hipStream_t stream) {
  (void)in_sizes; (void)n_in; (void)out_size; (void)ws_size;
  const float* X    = (const float*)d_in[0];
  const int*   lens = (const int*)d_in[1];
  const float* Wfw  = (const float*)d_in[2];
  const float* bfw  = (const float*)d_in[3];
  const float* Wbw  = (const float*)d_in[4];
  const float* bbw  = (const float*)d_in[5];
  const float* W1 = (const float*)d_in[6];  const float* b1 = (const float*)d_in[7];
  const float* W2 = (const float*)d_in[8];  const float* b2 = (const float*)d_in[9];
  const float* W3 = (const float*)d_in[10]; const float* b3 = (const float*)d_in[11];
  const float* W4 = (const float*)d_in[12]; const float* b4 = (const float*)d_in[13];

  char* ws = (char*)d_ws;
  int*    ctr  = (int*)(ws + WS_CTR_OFF);
  __bf16* hbuf = (__bf16*)(ws + WS_HBUF_OFF);
  __bf16* XP   = (__bf16*)(ws + WS_XP_OFF);
  __bf16* OUTb = (__bf16*)(ws + WS_OUT_OFF);
  float*  Y    = (float*)d_out;

  k0_init<<<dim3(20), dim3(256), 0, stream>>>(ctr);
  k1_xproj<<<dim3(256, 32, 10), dim3(256), 0, stream>>>(X, Wfw, bfw, Wbw, bbw, XP);
  (void)hipFuncSetAttribute((const void*)k2_lstm,
                            hipFuncAttributeMaxDynamicSharedMemorySize, K2_SMEM);
  k2_lstm<<<dim3(NDIR * NSPLIT), dim3(256), K2_SMEM, stream>>>(Wfw, Wbw, lens, XP, OUTb, hbuf, ctr);
  k3_heads<<<dim3(128, 4), dim3(256), 0, stream>>>(OUTb, W1, b1, W2, b2, W3, b3, W4, b4, Y);
}
